// TargetObsEncoderNoImage2HeadStdDescend_19456201851698
// MI455X (gfx1250) — matrix operand order repaired and checked by reading, not compiled, not run on hardware
//
#include <hip/hip_runtime.h>
#include <hip/hip_bf16.h>

// ---------------------------------------------------------------------------
// Types for CDNA5 WMMA (wave32, 16x16x32 bf16 -> f32)
// ---------------------------------------------------------------------------
typedef __bf16 bf16_t;
typedef bf16_t v16bf __attribute__((ext_vector_type(16)));
typedef bf16_t v8bf  __attribute__((ext_vector_type(8)));
typedef float  v8f   __attribute__((ext_vector_type(8)));

#define B_DIM   32
#define T_DIM   1024
#define NTOK    (B_DIM * T_DIM)

// ---------------------------------------------------------------------------
// Workspace layout (element offsets for bf16 transposed weights, [N,K] layout)
// ---------------------------------------------------------------------------
constexpr size_t OW0  = 0;                    // other0  96x512
constexpr size_t OW1  = OW0  + 96ull*512;     // other1  512x256
constexpr size_t OW2  = OW1  + 512ull*256;    // other2  256x32
constexpr size_t OW3  = OW2  + 256ull*32;     // pre0    32x1024
constexpr size_t OW4  = OW3  + 32ull*1024;    // pre1    1024x512
constexpr size_t OW5  = OW4  + 1024ull*512;   // pre2    512x256
constexpr size_t OW6  = OW5  + 512ull*256;    // Wih     256x768
constexpr size_t OW7  = OW6  + 256ull*768;    // Whh     256x768
constexpr size_t OW8  = OW7  + 256ull*768;    // post0   256x256
constexpr size_t OW9  = OW8  + 256ull*256;    // post1   256x128
constexpr size_t OW10 = OW9  + 256ull*128;    // mean0   128x512
constexpr size_t OW11 = OW10 + 128ull*512;    // mean1   512x256
constexpr size_t OW12 = OW11 + 512ull*256;    // mean2   256x64
constexpr size_t OW13 = OW12 + 256ull*64;     // logstd0 128x512
constexpr size_t OW14 = OW13 + 128ull*512;    // logstd1 512x256
constexpr size_t OW15 = OW14 + 512ull*256;    // logstd2 256x64

constexpr size_t XI_OFF  = 4ull << 20;                         // xi  f32 [NTOK,768]
constexpr size_t YSE_OFF = XI_OFF  + (size_t)NTOK*768*4;       // yse bf16 [NTOK,256]
constexpr size_t DL_OFF  = YSE_OFF + (size_t)NTOK*256*2;       // delta f32 [NTOK,64]

// ---------------------------------------------------------------------------
// Fragment loaders (wave32; A and B both use contiguous k = 16*(lane/16) + s, not the ISA table order)
// ---------------------------------------------------------------------------
__device__ inline v16bf loadA_lds(const bf16_t* act, int ldk, int mt, int kt, int lane) {
    int m  = lane & 15;
    int kb = (kt << 5) + ((lane >> 4) << 4);
    const bf16_t* p = act + (size_t)(mt * 16 + m) * ldk + kb;
    v8bf lo = *(const v8bf*)p;
    v8bf hi = *(const v8bf*)(p + 8);
    v16bf r;
#pragma unroll
    for (int i = 0; i < 8; ++i) { r[i] = lo[i]; r[i + 8] = hi[i]; }
    return r;
}

__device__ inline v16bf loadB_glb(const bf16_t* __restrict__ wt, int K, int nt, int kt, int lane) {
    int n  = (nt << 4) + (lane & 15);
    int kb = (kt << 5) + ((lane >> 4) << 4);
    return *(const v16bf*)(wt + (size_t)n * K + kb);
}

__device__ inline float elu_f(float x) { return x > 0.f ? x : __expf(x) - 1.f; }
__device__ inline float sigmoid_f(float x) { return 1.f / (1.f + __expf(-x)); }

// ---------------------------------------------------------------------------
// Fused GEMM layer with 4-way M-blocking: each wave owns one n-tile and runs
// all 4 m-tiles against a single B fragment (4 WMMA per weight load).
// LDS(bf16 [64,K]) x W^T(bf16 [N,K]) -> LDS(bf16 [64,N])
// ---------------------------------------------------------------------------
__device__ inline void gemm_layer(const bf16_t* __restrict__ wt, const float* __restrict__ bias,
                                  const bf16_t* in, int K, bf16_t* out, int N,
                                  int wave, int nwaves, int lane, bool do_elu) {
    int ntN = N >> 4;
    int KT  = K >> 5;
    for (int nt = wave; nt < ntN; nt += nwaves) {
        int n  = (nt << 4) + (lane & 15);
        float bv = bias[n];
        v8f a0, a1, a2, a3;
#pragma unroll
        for (int i = 0; i < 8; ++i) { a0[i] = bv; a1[i] = bv; a2[i] = bv; a3[i] = bv; }
        for (int kt = 0; kt < KT; ++kt) {
            v16bf b = loadB_glb(wt, K, nt, kt, lane);
            a0 = __builtin_amdgcn_wmma_f32_16x16x32_bf16(false, loadA_lds(in, K, 0, kt, lane),
                                                         false, b, (short)0, a0, false, false);
            a1 = __builtin_amdgcn_wmma_f32_16x16x32_bf16(false, loadA_lds(in, K, 1, kt, lane),
                                                         false, b, (short)0, a1, false, false);
            a2 = __builtin_amdgcn_wmma_f32_16x16x32_bf16(false, loadA_lds(in, K, 2, kt, lane),
                                                         false, b, (short)0, a2, false, false);
            a3 = __builtin_amdgcn_wmma_f32_16x16x32_bf16(false, loadA_lds(in, K, 3, kt, lane),
                                                         false, b, (short)0, a3, false, false);
        }
        int mb = (lane >> 4) << 3;
#pragma unroll
        for (int v = 0; v < 8; ++v) {
            float x0 = a0[v], x1 = a1[v], x2 = a2[v], x3 = a3[v];
            if (do_elu) { x0 = elu_f(x0); x1 = elu_f(x1); x2 = elu_f(x2); x3 = elu_f(x3); }
            out[(size_t)(mb + v) * N + n]      = (bf16_t)x0;
            out[(size_t)(16 + mb + v) * N + n] = (bf16_t)x1;
            out[(size_t)(32 + mb + v) * N + n] = (bf16_t)x2;
            out[(size_t)(48 + mb + v) * N + n] = (bf16_t)x3;
        }
    }
}

// Tile-per-wave variant storing f32 to global (better balance for small N heads)
// mode: 0=none, 1=elu, 2=-tanh
__device__ inline void gemm_out(const bf16_t* __restrict__ wt, const float* __restrict__ bias,
                                const bf16_t* in, int K, float* __restrict__ outg,
                                int ldo, int N, int MT, long tokBase,
                                int wave, int nwaves, int lane, int mode) {
    int ntN = N >> 4;
    int KT  = K >> 5;
    for (int tile = wave; tile < MT * ntN; tile += nwaves) {
        int mt = tile / ntN, nt = tile % ntN;
        int n  = (nt << 4) + (lane & 15);
        float bv = bias[n];
        v8f acc;
#pragma unroll
        for (int i = 0; i < 8; ++i) acc[i] = bv;
        for (int kt = 0; kt < KT; ++kt) {
            v16bf a = loadA_lds(in, K, mt, kt, lane);
            v16bf b = loadB_glb(wt, K, nt, kt, lane);
            acc = __builtin_amdgcn_wmma_f32_16x16x32_bf16(false, a, false, b,
                                                          (short)0, acc, false, false);
        }
        int mbase = mt * 16 + ((lane >> 4) << 3);
#pragma unroll
        for (int v = 0; v < 8; ++v) {
            float x = acc[v];
            if (mode == 1) x = elu_f(x);
            else if (mode == 2) x = -tanhf(x);
            outg[(tokBase + mbase + v) * (long)ldo + n] = x;
        }
    }
}

// ---------------------------------------------------------------------------
// Kernel 0: f32 -> bf16 transpose of a weight matrix (W[K,N] -> Wt[N,K])
// ---------------------------------------------------------------------------
__global__ void wconv(const float* __restrict__ src, bf16_t* __restrict__ dst, int K, int N) {
    int i = blockIdx.x * 256 + threadIdx.x;
    if (i >= K * N) return;
    int n = i / K, k = i % K;
    dst[i] = (bf16_t)src[(size_t)k * N + n];
}

// ---------------------------------------------------------------------------
// Kernel 1: token-parallel chain  concat(x,la) ->512->256->32->1024->512->256 -> xi(768)
// ---------------------------------------------------------------------------
__global__ __launch_bounds__(512) void phase1(
    const float* __restrict__ x, const float* __restrict__ la,
    const float* __restrict__ bo0, const float* __restrict__ bo1, const float* __restrict__ bo2,
    const float* __restrict__ bp0, const float* __restrict__ bp1, const float* __restrict__ bp2,
    const float* __restrict__ bih, const bf16_t* __restrict__ wt, float* __restrict__ xi) {
    extern __shared__ bf16_t smem1[];
    bf16_t* buf0 = smem1;                 // 64*1024 elements
    bf16_t* buf1 = smem1 + 64 * 1024;     // 64*512 elements
    int tid = threadIdx.x, lane = tid & 31, wave = tid >> 5;
    long tokBase = (long)blockIdx.x * 64;

    for (int i = tid; i < 64 * 96; i += 512) {
        int r = i / 96, f = i % 96;
        long tok = tokBase + r;
        float v = (f < 80) ? x[tok * 80 + f] : la[tok * 16 + (f - 80)];
        buf0[(size_t)r * 96 + f] = (bf16_t)v;
    }
    __syncthreads();
    gemm_layer(wt + OW0, bo0, buf0,   96, buf1,  512, wave, 16, lane, true);
    __syncthreads();
    gemm_layer(wt + OW1, bo1, buf1,  512, buf0,  256, wave, 16, lane, true);
    __syncthreads();
    gemm_layer(wt + OW2, bo2, buf0,  256, buf1,   32, wave, 16, lane, true);
    __syncthreads();
    gemm_layer(wt + OW3, bp0, buf1,   32, buf0, 1024, wave, 16, lane, true);
    __syncthreads();
    gemm_layer(wt + OW4, bp1, buf0, 1024, buf1,  512, wave, 16, lane, true);
    __syncthreads();
    gemm_layer(wt + OW5, bp2, buf1,  512, buf0,  256, wave, 16, lane, true);
    __syncthreads();
    gemm_out(wt + OW6, bih, buf0, 256, xi, 768, 768, 4, tokBase, wave, 16, lane, 0);
}

// ---------------------------------------------------------------------------
// Kernel 2: serial GRU scan. 2 blocks x 16 batch rows, 16 waves each.
// The wave's Whh slice (3 n-tiles x 8 k-tiles = 192 VGPRs) is hoisted into
// registers before the 1024-step loop; A-fragments hoisted per step; episode
// reset fused into the gate phase -> 2 barriers/step and zero weight traffic
// on the serial critical path.
// ---------------------------------------------------------------------------
__global__ __launch_bounds__(512) void gru_scan(
    const bf16_t* __restrict__ wtWhh, const float* __restrict__ bhh,
    const float* __restrict__ xi, const int* __restrict__ rs,
    const float* __restrict__ h0, bf16_t* __restrict__ yse) {
    extern __shared__ unsigned char smem2[];
    float*  h32 = (float*)smem2;              // 16*256 f32
    float*  hh  = h32 + 16 * 256;             // 16*768 f32
    bf16_t* hbf = (bf16_t*)(hh + 16 * 768);   // 16*256 bf16
    int tid = threadIdx.x, lane = tid & 31, wave = tid >> 5;
    int bbase = blockIdx.x * 16;

    // ---- hoist the wave's Whh slice into registers (24 fragments) ----
    v16bf Bf[3][8];
#pragma unroll
    for (int i = 0; i < 3; ++i)
#pragma unroll
        for (int kt = 0; kt < 8; ++kt)
            Bf[i][kt] = loadB_glb(wtWhh, 256, wave * 3 + i, kt, lane);
    float bvv[3];
#pragma unroll
    for (int i = 0; i < 3; ++i)
        bvv[i] = bhh[(wave * 3 + i) * 16 + (lane & 15)];

    // ---- init h with the t=0 reset already applied ----
    for (int i = tid; i < 16 * 256; i += 512) {
        int r = i >> 8;
        float s = (float)rs[(size_t)(bbase + r) * T_DIM + 0];
        float hv = h0[(size_t)(bbase + r) * 256 + (i & 255)] * (1.f - s);
        h32[i] = hv;
        hbf[i] = (bf16_t)hv;
    }

    for (int t = 0; t < T_DIM; ++t) {
        __syncthreads();   // hbf (post-reset h) ready for all waves
        // ---- hh = h @ Whh + bhh  (M=16, N=768, K=256) ----
        v16bf Af[8];
#pragma unroll
        for (int kt = 0; kt < 8; ++kt)
            Af[kt] = loadA_lds(hbf, 256, 0, kt, lane);
#pragma unroll
        for (int i = 0; i < 3; ++i) {
            v8f acc;
#pragma unroll
            for (int v = 0; v < 8; ++v) acc[v] = bvv[i];
#pragma unroll
            for (int kt = 0; kt < 8; ++kt)
                acc = __builtin_amdgcn_wmma_f32_16x16x32_bf16(false, Af[kt], false, Bf[i][kt],
                                                              (short)0, acc, false, false);
            int n  = (wave * 3 + i) * 16 + (lane & 15);
            int mb = (lane >> 4) << 3;
#pragma unroll
            for (int v = 0; v < 8; ++v)
                hh[(size_t)(mb + v) * 768 + n] = acc[v];
        }
        __syncthreads();   // hh ready
        // ---- gates + fused reset for step t+1 ----
#pragma unroll
        for (int u = 0; u < 8; ++u) {
            int idx = tid * 8 + u;
            int r = idx >> 8, j = idx & 255;
            const float* xib = xi + ((size_t)(bbase + r) * T_DIM + t) * 768;
            const float* hhr = hh + (size_t)r * 768;
            float rg = sigmoid_f(xib[j]       + hhr[j]);
            float zg = sigmoid_f(xib[256 + j] + hhr[256 + j]);
            float nn = tanhf(xib[512 + j] + rg * hhr[512 + j]);
            float hn = (1.f - zg) * nn + zg * h32[idx];
            yse[((size_t)(bbase + r) * T_DIM + t) * 256 + j] = (bf16_t)elu_f(hn);
            float s = (t + 1 < T_DIM) ? (float)rs[(size_t)(bbase + r) * T_DIM + t + 1] : 0.f;
            float hv = hn * (1.f - s);
            h32[idx] = hv;
            hbf[idx] = (bf16_t)hv;
        }
    }
}

// ---------------------------------------------------------------------------
// Kernel 3: token-parallel post/mean/logstd. 64 tokens/block, 16 waves.
// ---------------------------------------------------------------------------
__global__ __launch_bounds__(512) void phase3(
    const bf16_t* __restrict__ yse,
    const float* __restrict__ bq0, const float* __restrict__ bq1,
    const float* __restrict__ bm0, const float* __restrict__ bm1, const float* __restrict__ bm2,
    const float* __restrict__ bl0, const float* __restrict__ bl1, const float* __restrict__ bl2,
    const bf16_t* __restrict__ wt, float* __restrict__ out, float* __restrict__ delta) {
    extern __shared__ bf16_t smem3[];
    bf16_t* s_in  = smem3;               // 64*256
    bf16_t* s_t0  = s_in  + 64 * 256;    // 64*256
    bf16_t* s_emb = s_t0  + 64 * 256;    // 64*128
    bf16_t* s_t1  = s_emb + 64 * 128;    // 64*512
    bf16_t* s_t2  = s_t1  + 64 * 512;    // 64*256
    int tid = threadIdx.x, lane = tid & 31, wave = tid >> 5;
    long tokBase = (long)blockIdx.x * 64;

    // vectorized stage of yse tile (32KB) into LDS
    {
        const uint4* src = (const uint4*)(yse + tokBase * 256);
        uint4* dst = (uint4*)s_in;
        for (int i = tid; i < (64 * 256) / 8; i += 512) dst[i] = src[i];
    }
    __syncthreads();
    gemm_layer(wt + OW8,  bq0, s_in,  256, s_t0,  256, wave, 16, lane, true);
    __syncthreads();
    gemm_layer(wt + OW9,  bq1, s_t0,  256, s_emb, 128, wave, 16, lane, true);
    __syncthreads();
    // mean head
    gemm_layer(wt + OW10, bm0, s_emb, 128, s_t1,  512, wave, 16, lane, true);
    __syncthreads();
    gemm_layer(wt + OW11, bm1, s_t1,  512, s_t2,  256, wave, 16, lane, true);
    __syncthreads();
    gemm_out(wt + OW12, bm2, s_t2, 256, out, 128, 64, 4, tokBase, wave, 16, lane, 1);
    __syncthreads();
    // logstd head (delta = -tanh(mlp))
    gemm_layer(wt + OW13, bl0, s_emb, 128, s_t1,  512, wave, 16, lane, true);
    __syncthreads();
    gemm_layer(wt + OW14, bl1, s_t1,  512, s_t2,  256, wave, 16, lane, true);
    __syncthreads();
    gemm_out(wt + OW15, bl2, s_t2, 256, delta, 64, 64, 4, tokBase, wave, 16, lane, 2);
}

// ---------------------------------------------------------------------------
// Kernel 4: segment cumsum of delta_logstd over T, per (b, e) lane.
// ---------------------------------------------------------------------------
__global__ void seg_cumsum(const float* __restrict__ delta, const int* __restrict__ rs,
                           const float* __restrict__ hls0, float* __restrict__ out) {
    int gid = blockIdx.x * blockDim.x + threadIdx.x;
    if (gid >= B_DIM * 64) return;
    int b = gid >> 6, e = gid & 63;
    float c = hls0[(size_t)b * 64 + e];
    for (int t = 0; t < T_DIM; ++t) {
        float s = (float)rs[(size_t)b * T_DIM + t];
        float d = delta[((size_t)b * T_DIM + t) * 64 + e];
        c = d + c * (1.f - s);
        out[((size_t)b * T_DIM + t) * 128 + 64 + e] = c;
    }
}

// ---------------------------------------------------------------------------
// Host launcher
// ---------------------------------------------------------------------------
extern "C" void kernel_launch(void* const* d_in, const int* in_sizes, int n_in,
                              void* d_out, int out_size, void* d_ws, size_t ws_size,
                              hipStream_t stream) {
    const float* x    = (const float*)d_in[0];
    const float* la   = (const float*)d_in[1];
    const int*   rs   = (const int*)d_in[2];
    const float* hls0 = (const float*)d_in[3];
    const float* hg0  = (const float*)d_in[4];

    bf16_t* wt    = (bf16_t*)d_ws;
    float*  xi    = (float*)((char*)d_ws + XI_OFF);
    bf16_t* yse   = (bf16_t*)((char*)d_ws + YSE_OFF);
    float*  delta = (float*)((char*)d_ws + DL_OFF);
    float*  out   = (float*)d_out;

    struct WSpec { int src; size_t off; int K, N; };
    const WSpec ws[16] = {
        {5,  OW0,   96,  512}, {7,  OW1,  512, 256}, {9,  OW2,  256,  32},
        {11, OW3,   32, 1024}, {13, OW4, 1024, 512}, {15, OW5,  512, 256},
        {17, OW6,  256,  768}, {18, OW7,  256, 768},
        {21, OW8,  256,  256}, {23, OW9,  256, 128},
        {25, OW10, 128,  512}, {27, OW11, 512, 256}, {29, OW12, 256,  64},
        {31, OW13, 128,  512}, {33, OW14, 512, 256}, {35, OW15, 256,  64},
    };
    for (int i = 0; i < 16; ++i) {
        int total = ws[i].K * ws[i].N;
        wconv<<<(total + 255) / 256, 256, 0, stream>>>(
            (const float*)d_in[ws[i].src], wt + ws[i].off, ws[i].K, ws[i].N);
    }

    phase1<<<NTOK / 64, 512, (64 * 1024 + 64 * 512) * sizeof(bf16_t), stream>>>(
        x, la,
        (const float*)d_in[6],  (const float*)d_in[8],  (const float*)d_in[10],
        (const float*)d_in[12], (const float*)d_in[14], (const float*)d_in[16],
        (const float*)d_in[19], wt, xi);

    size_t smem2 = (16 * 256 + 16 * 768) * sizeof(float) + 16 * 256 * sizeof(bf16_t);
    gru_scan<<<2, 512, smem2, stream>>>(wt + OW7, (const float*)d_in[20], xi, rs, hg0, yse);

    size_t smem3 = (64 * 256 + 64 * 256 + 64 * 128 + 64 * 512 + 64 * 256) * sizeof(bf16_t);
    phase3<<<NTOK / 64, 512, smem3, stream>>>(
        yse,
        (const float*)d_in[22], (const float*)d_in[24],
        (const float*)d_in[26], (const float*)d_in[28], (const float*)d_in[30],
        (const float*)d_in[32], (const float*)d_in[34], (const float*)d_in[36],
        wt, out, delta);

    seg_cumsum<<<(B_DIM * 64 + 255) / 256, 256, 0, stream>>>(delta, rs, hls0, out);
}
